// RecurrentSimplE_89077621719318
// MI455X (gfx1250) — matrix operand order repaired and checked by reading, not compiled, not run on hardware
//
#include <hip/hip_runtime.h>

typedef __attribute__((ext_vector_type(16))) __bf16 v16bf;
typedef __attribute__((ext_vector_type(8)))  __bf16 v8bf;
typedef __attribute__((ext_vector_type(8)))  float  v8f;

#define SQ   12
#define BB   512
#define EE   128
#define HH   8
#define FF   512
#define NENT 100000
#define TOK  (SQ*BB)   // 6144 token rows

union V16U { v16bf v; v8bf h[2]; };

// ---------------------------------------------------------------- f32 -> bf16
__global__ void cvt_f32_bf16(const float* __restrict__ src, __bf16* __restrict__ dst, int n) {
  int i = blockIdx.x * blockDim.x + threadIdx.x;
  if (i < n) dst[i] = (__bf16)src[i];
}

// ------------------------------------------------------------- build sequence
__global__ void gather_seq(const int* __restrict__ inp,
                           const float* __restrict__ subj,
                           const float* __restrict__ ent,  const float* __restrict__ entt,
                           const float* __restrict__ rel,  const float* __restrict__ relinv,
                           const float* __restrict__ ctx,
                           float* __restrict__ xf, __bf16* __restrict__ xb) {
  int idx = blockIdx.x * blockDim.x + threadIdx.x;   // TOK*EE threads
  int d = idx & (EE - 1);
  int t = idx >> 7;            // token row = s*BB + b
  int b = t & (BB - 1);
  int s = t >> 9;
  const float* src;
  if      (s == 0)  src = ent    + (size_t)inp[0*BB + b] * EE;
  else if (s == 1)  src = entt   + (size_t)inp[0*BB + b] * EE;
  else if (s == 2)  src = rel    + (size_t)inp[1*BB + b] * EE;
  else if (s == 3)  src = relinv + (size_t)inp[1*BB + b] * EE;
  else if (s == 4)  src = ent    + (size_t)inp[2*BB + b] * EE;
  else if (s == 5)  src = entt   + (size_t)inp[2*BB + b] * EE;
  else if (s <= 10) src = ctx    + (size_t)inp[(3 + (s - 6))*BB + b] * EE;
  else              src = subj   + (size_t)b * EE;
  float v = src[d];
  xf[idx] = v;
  xb[idx] = (__bf16)v;
}

// --------------------------------------------------- generic GEMM  C = A @ B^T
// A: bf16 [M,K] row-major, B: bf16 [N,K] row-major (so per-lane K-contiguous),
// C: f32 [M,N] (optional), Cb: bf16 [M,N] (optional). block=256 thr (8 waves),
// wave tile 16Mx64N, block tile 64Mx128N. M%64==0, N%128==0, K%32==0.
__global__ void gemm_bf16nt(const __bf16* __restrict__ A, const __bf16* __restrict__ Bm,
                            const float* __restrict__ bias,
                            float* __restrict__ C, __bf16* __restrict__ Cb,
                            int M, int N, int K, int relu) {
  int lane = threadIdx.x & 31;
  int wave = threadIdx.x >> 5;
  int ln = lane & 15;
  int hi = lane >> 4;
  int M0 = blockIdx.x * 64  + (wave & 3) * 16;
  int N0 = blockIdx.y * 128 + (wave >> 2) * 64;
  int m  = M0 + ln;

  v8f acc[4];
#pragma unroll
  for (int j = 0; j < 4; ++j)
#pragma unroll
    for (int i = 0; i < 8; ++i) acc[j][i] = 0.0f;

  for (int k0 = 0; k0 < K; k0 += 32) {
    V16U a;
    const __bf16* pa = A + (size_t)m * K + k0 + hi * 16;
    a.h[0] = *(const v8bf*)(pa);
    a.h[1] = *(const v8bf*)(pa + 8);
#pragma unroll
    for (int j = 0; j < 4; ++j) {
      int n = N0 + j * 16 + ln;
      v16bf bf = *(const v16bf*)(Bm + (size_t)n * K + k0 + hi * 16);
      acc[j] = __builtin_amdgcn_wmma_f32_16x16x32_bf16(false, a.v, false, bf,
                                                       (short)0, acc[j], false, false);
    }
  }

  int mrow = M0 + hi * 8;
#pragma unroll
  for (int j = 0; j < 4; ++j) {
    int n = N0 + j * 16 + ln;
    float bv = bias ? bias[n] : 0.0f;
#pragma unroll
    for (int r = 0; r < 8; ++r) {
      float v = acc[j][r] + bv;
      if (relu) v = fmaxf(v, 0.0f);
      size_t o = (size_t)(mrow + r) * N + n;
      if (C)  C[o]  = v;
      if (Cb) Cb[o] = (__bf16)v;
    }
  }
}

// ---------------------------------------------------------- tiny attention
// one thread per (b,h,s); S=12, Dh=16
__global__ void attn_kernel(const float* __restrict__ qkv, __bf16* __restrict__ ob) {
  int gid = blockIdx.x * blockDim.x + threadIdx.x;  // BB*HH*SQ threads
  int s = gid % SQ;
  int h = (gid / SQ) % HH;
  int b = gid / (SQ * HH);
  const float scale = 0.25f;  // 1/sqrt(16)
  const float* qr = qkv + (size_t)(s * BB + b) * 384 + h * 16;
  float q[16];
#pragma unroll
  for (int d = 0; d < 16; ++d) q[d] = qr[d];
  float att[SQ];
  float mx = -1e30f;
  for (int t = 0; t < SQ; ++t) {
    const float* kr = qkv + (size_t)(t * BB + b) * 384 + 128 + h * 16;
    float dot = 0.f;
#pragma unroll
    for (int d = 0; d < 16; ++d) dot += q[d] * kr[d];
    dot *= scale;
    att[t] = dot;
    mx = fmaxf(mx, dot);
  }
  float sum = 0.f;
  for (int t = 0; t < SQ; ++t) { att[t] = __expf(att[t] - mx); sum += att[t]; }
  float inv = 1.0f / sum;
  float o[16];
#pragma unroll
  for (int d = 0; d < 16; ++d) o[d] = 0.f;
  for (int t = 0; t < SQ; ++t) {
    const float* vr = qkv + (size_t)(t * BB + b) * 384 + 256 + h * 16;
    float wgt = att[t] * inv;
#pragma unroll
    for (int d = 0; d < 16; ++d) o[d] += wgt * vr[d];
  }
  __bf16* orow = ob + (size_t)(s * BB + b) * EE + h * 16;
#pragma unroll
  for (int d = 0; d < 16; ++d) orow[d] = (__bf16)o[d];
}

// --------------------------------------------- residual + LayerNorm (1 wave/row)
__global__ void ln_residual(float* __restrict__ x, const float* __restrict__ dlt,
                            const float* __restrict__ g, const float* __restrict__ be,
                            __bf16* __restrict__ xb, int rows) {
  int wave = threadIdx.x >> 5, lane = threadIdx.x & 31;
  int row = blockIdx.x * 8 + wave;
  if (row >= rows) return;
  float* xr = x + (size_t)row * EE;
  const float* dr = dlt + (size_t)row * EE;
  float v[4];
  float s = 0.f;
#pragma unroll
  for (int i = 0; i < 4; ++i) { v[i] = xr[lane * 4 + i] + dr[lane * 4 + i]; s += v[i]; }
#pragma unroll
  for (int off = 16; off; off >>= 1) s += __shfl_xor(s, off, 32);
  float mu = s * (1.0f / EE);
  float var = 0.f;
#pragma unroll
  for (int i = 0; i < 4; ++i) { float d = v[i] - mu; var += d * d; }
#pragma unroll
  for (int off = 16; off; off >>= 1) var += __shfl_xor(var, off, 32);
  var *= (1.0f / EE);
  float r = rsqrtf(var + 1e-5f);
  __bf16* xbr = xb + (size_t)row * EE;
#pragma unroll
  for (int i = 0; i < 4; ++i) {
    int c = lane * 4 + i;
    float y = (v[i] - mu) * r * g[c] + be[c];
    xr[c] = y;
    xbr[c] = (__bf16)y;
  }
}

// -------------------------------------- build fused query matrix Q = [U | W] bf16
__global__ void build_q(const float* __restrict__ x, const int* __restrict__ nxt,
                        const float* __restrict__ rel, const float* __restrict__ relinv,
                        __bf16* __restrict__ Q) {
  int idx = blockIdx.x * blockDim.x + threadIdx.x;   // BB*EE
  int d = idx & (EE - 1);
  int b = idx >> 7;
  int r = nxt[1 * BB + b];
  float u = x[(size_t)(0 * BB + b) * EE + d] * rel[(size_t)r * EE + d];
  float w = x[(size_t)(1 * BB + b) * EE + d] * relinv[(size_t)r * EE + d];
  Q[(size_t)b * 256 + d]       = (__bf16)u;
  Q[(size_t)b * 256 + 128 + d] = (__bf16)w;
}

__global__ void copy_history(const float* __restrict__ x, float* __restrict__ out) {
  int idx = blockIdx.x * blockDim.x + threadIdx.x;   // BB*EE
  int d = idx & (EE - 1);
  int b = idx >> 7;
  out[(size_t)NENT * BB + idx] = x[(size_t)(11 * BB + b) * EE + d];
}

// ------------------- fused scoring GEMM: out = clip(0.5 * Q @ [ent|enttail]^T)
// M=512, N=100000, K=256. B-side read fp32, converted to bf16 in-register.
__global__ void score_gemm(const __bf16* __restrict__ Q,
                           const float* __restrict__ ent, const float* __restrict__ entt,
                           float* __restrict__ out) {
  int lane = threadIdx.x & 31, wave = threadIdx.x >> 5;
  int ln = lane & 15, hi = lane >> 4;
  int M0 = blockIdx.x * 64  + (wave & 3) * 16;
  int N0 = blockIdx.y * 128 + (wave >> 2) * 64;
  int m  = M0 + ln;

  v8f acc[4];
#pragma unroll
  for (int j = 0; j < 4; ++j)
#pragma unroll
    for (int i = 0; i < 8; ++i) acc[j][i] = 0.0f;

  for (int k0 = 0; k0 < 256; k0 += 32) {
    V16U a;
    const __bf16* pa = Q + (size_t)m * 256 + k0 + hi * 16;
    a.h[0] = *(const v8bf*)(pa);
    a.h[1] = *(const v8bf*)(pa + 8);
    int kk = k0 + hi * 16;                      // 16 contiguous K values per lane
    const float* tbl = (kk < 128) ? ent : entt; // fused [ent | enttail] K-space
    int kc = kk & 127;
#pragma unroll
    for (int j = 0; j < 4; ++j) {
      int n = N0 + j * 16 + ln;
      if (n >= NENT) n = NENT - 1;              // clamp loads at N edge (store guarded)
      const float* pf = tbl + (size_t)n * EE + kc;
      v16bf bf;
#pragma unroll
      for (int i = 0; i < 16; ++i) bf[i] = (__bf16)pf[i];
      acc[j] = __builtin_amdgcn_wmma_f32_16x16x32_bf16(false, a.v, false, bf,
                                                       (short)0, acc[j], false, false);
    }
  }

  int mrow = M0 + hi * 8;
#pragma unroll
  for (int j = 0; j < 4; ++j) {
    int n = N0 + j * 16 + ln;
    if (n < NENT) {
#pragma unroll
      for (int r = 0; r < 8; ++r) {
        float v = acc[j][r] * 0.5f;
        v = fminf(fmaxf(v, -20.0f), 20.0f);
        out[(size_t)(mrow + r) * NENT + n] = v;
      }
    }
  }
}

// ------------------------------------------------------------------- launcher
extern "C" void kernel_launch(void* const* d_in, const int* in_sizes, int n_in,
                              void* d_out, int out_size, void* d_ws, size_t ws_size,
                              hipStream_t stream) {
  (void)in_sizes; (void)n_in; (void)out_size; (void)ws_size;
  const int*   inp    = (const int*)d_in[0];
  const int*   nxt    = (const int*)d_in[1];
  const float* subj   = (const float*)d_in[2];
  const float* ent    = (const float*)d_in[3];
  const float* entt   = (const float*)d_in[4];
  const float* rel    = (const float*)d_in[5];
  const float* relinv = (const float*)d_in[6];
  const float* ctx    = (const float*)d_in[7];
  const float* Wqkv   = (const float*)d_in[8];
  const float* bqkv   = (const float*)d_in[9];
  const float* Wo     = (const float*)d_in[10];
  const float* bo     = (const float*)d_in[11];
  const float* W1     = (const float*)d_in[12];
  const float* b1     = (const float*)d_in[13];
  const float* W2     = (const float*)d_in[14];
  const float* b2     = (const float*)d_in[15];
  const float* ln1g   = (const float*)d_in[16];
  const float* ln1b   = (const float*)d_in[17];
  const float* ln2g   = (const float*)d_in[18];
  const float* ln2b   = (const float*)d_in[19];
  float* out = (float*)d_out;

  char* w = (char*)d_ws;
  size_t off = 0;
  auto alloc = [&](size_t bytes) -> char* {
    char* p = w + off;
    off = (off + bytes + 255) & ~(size_t)255;
    return p;
  };
  float*  xf    = (float*) alloc((size_t)TOK * EE * 4);
  __bf16* xb    = (__bf16*)alloc((size_t)TOK * EE * 2);
  float*  qkv   = (float*) alloc((size_t)TOK * 384 * 4);
  __bf16* ob    = (__bf16*)alloc((size_t)TOK * EE * 2);
  float*  t1    = (float*) alloc((size_t)TOK * EE * 4);
  __bf16* hb    = (__bf16*)alloc((size_t)TOK * FF * 2);
  __bf16* Wqkvb = (__bf16*)alloc((size_t)2 * 384 * EE * 2);
  __bf16* Wob   = (__bf16*)alloc((size_t)2 * EE * EE * 2);
  __bf16* W1b   = (__bf16*)alloc((size_t)2 * FF * EE * 2);
  __bf16* W2b   = (__bf16*)alloc((size_t)2 * EE * FF * 2);
  __bf16* Qb    = (__bf16*)alloc((size_t)BB * 256 * 2);

  // static weight conversions to bf16
  cvt_f32_bf16<<<(2*384*EE + 255) / 256, 256, 0, stream>>>(Wqkv, Wqkvb, 2*384*EE);
  cvt_f32_bf16<<<(2*EE*EE  + 255) / 256, 256, 0, stream>>>(Wo,   Wob,   2*EE*EE);
  cvt_f32_bf16<<<(2*FF*EE  + 255) / 256, 256, 0, stream>>>(W1,   W1b,   2*FF*EE);
  cvt_f32_bf16<<<(2*EE*FF  + 255) / 256, 256, 0, stream>>>(W2,   W2b,   2*EE*FF);

  gather_seq<<<TOK * EE / 256, 256, 0, stream>>>(inp, subj, ent, entt, rel, relinv, ctx, xf, xb);

  for (int l = 0; l < 2; ++l) {
    dim3 g1(TOK / 64, 384 / 128);
    gemm_bf16nt<<<g1, 256, 0, stream>>>(xb, Wqkvb + (size_t)l*384*EE, bqkv + l*384,
                                        qkv, nullptr, TOK, 384, EE, 0);
    attn_kernel<<<BB * HH * SQ / 256, 256, 0, stream>>>(qkv, ob);
    dim3 g2(TOK / 64, 1);
    gemm_bf16nt<<<g2, 256, 0, stream>>>(ob, Wob + (size_t)l*EE*EE, bo + l*EE,
                                        t1, nullptr, TOK, EE, EE, 0);
    ln_residual<<<TOK / 8, 256, 0, stream>>>(xf, t1, ln1g + l*EE, ln1b + l*EE, xb, TOK);
    dim3 g3(TOK / 64, FF / 128);
    gemm_bf16nt<<<g3, 256, 0, stream>>>(xb, W1b + (size_t)l*FF*EE, b1 + l*FF,
                                        nullptr, hb, TOK, FF, EE, 1);
    dim3 g4(TOK / 64, 1);
    gemm_bf16nt<<<g4, 256, 0, stream>>>(hb, W2b + (size_t)l*EE*FF, b2 + l*EE,
                                        t1, nullptr, TOK, EE, FF, 0);
    ln_residual<<<TOK / 8, 256, 0, stream>>>(xf, t1, ln2g + l*EE, ln2b + l*EE, xb, TOK);
  }

  copy_history<<<BB * EE / 256, 256, 0, stream>>>(xf, out);
  build_q<<<BB * EE / 256, 256, 0, stream>>>(xf, nxt, rel, relinv, Qb);
  dim3 gs(BB / 64, (NENT + 127) / 128);
  score_gemm<<<gs, 256, 0, stream>>>(Qb, ent, entt, out);
}
